// FlexOlmoDecoderLayer_4054449127760
// MI455X (gfx1250) — compile-verified, matrix operand order checked by reading, not run on hardware
//
#include <hip/hip_runtime.h>
#include <hip/hip_bf16.h>
#include <math.h>

#define TT 2048
#define HIDDEN 2048
#define NH 16
#define HD 128
#define NEXP 8
#define INTER 1024
#define EPSV 1e-5f

typedef __attribute__((ext_vector_type(16))) __bf16 v16bf;
typedef __attribute__((ext_vector_type(8)))  float  v8f;
typedef __attribute__((ext_vector_type(4)))  unsigned int u32x4;
typedef __attribute__((ext_vector_type(8)))  unsigned int u32x8;

union Frag { v16bf v; uint4 q[2]; };

__device__ __forceinline__ unsigned short f2bf(float f) {
    unsigned int u = __float_as_uint(f);
    u += 0x7FFFu + ((u >> 16) & 1u);   // round-to-nearest-even
    return (unsigned short)(u >> 16);
}

__device__ __forceinline__ v8f wmma_bf16(const v16bf& a, const v16bf& b, const v8f& c) {
    return __builtin_amdgcn_wmma_f32_16x16x32_bf16(false, a, false, b, (short)0, c, false, false);
}

// Load one 16x32 bf16 operand fragment (A-layout): lane r (=lane&15) carries row r;
// half (=lane>>4) selects K-subwindow {half*8..+7, 16+half*8..+7} as two 16B chunks.
__device__ __forceinline__ Frag load2(const unsigned short* p) {
    Frag f;
    f.q[0] = *(const uint4*)(p);
    f.q[1] = *(const uint4*)(p + 16);
    return f;
}

// ---------------------------------------------------------------------------
// Generic bf16 WMMA GEMM: C[M,N] = A[M,K] @ B[N,K]^T   (A,B bf16 bits, C fp32)
// mode 0: C = val      mode 1: C += rowscale[row*NEXP + expert] * val
// Block: 256 thr (8 waves, 4x2), block tile 128x128, wave tile 32x64, K-step 32.
// ---------------------------------------------------------------------------
__global__ __launch_bounds__(256) void gemm_bf16(
    const unsigned short* __restrict__ A, const unsigned short* __restrict__ B,
    float* __restrict__ C, int M, int N, int K,
    int mode, const float* __restrict__ rowscale, int expert)
{
    const int tid  = threadIdx.x;
    const int wave = tid >> 5;
    const int lane = tid & 31;
    const int half = lane >> 4;
    const int r    = lane & 15;
    const int wm   = wave >> 1;                 // 0..3
    const int wn   = wave & 1;                  // 0..1
    const int m_base = blockIdx.y * 128 + wm * 32;
    const int n_base = blockIdx.x * 128 + wn * 64;

    v8f acc[2][4];
#pragma unroll
    for (int mt = 0; mt < 2; ++mt)
#pragma unroll
        for (int nt = 0; nt < 4; ++nt)
#pragma unroll
            for (int i = 0; i < 8; ++i) acc[mt][nt][i] = 0.0f;

    for (int k0 = 0; k0 < K; k0 += 32) {
        Frag a[2], b[4];
#pragma unroll
        for (int mt = 0; mt < 2; ++mt)
            a[mt] = load2(A + (size_t)(m_base + mt * 16 + r) * K + k0 + half * 8);
#pragma unroll
        for (int nt = 0; nt < 4; ++nt)
            b[nt] = load2(B + (size_t)(n_base + nt * 16 + r) * K + k0 + half * 8);
        // Unconditional speculative prefetch of the next K-tile (branch-free:
        // a translation miss past the last tile is silently dropped).
        __builtin_prefetch(A + (size_t)(m_base + r) * K + k0 + 32, 0, 1);
        __builtin_prefetch(B + (size_t)(n_base + r) * K + k0 + 32, 0, 1);
#pragma unroll
        for (int mt = 0; mt < 2; ++mt)
#pragma unroll
            for (int nt = 0; nt < 4; ++nt)
                acc[mt][nt] = wmma_bf16(a[mt].v, b[nt].v, acc[mt][nt]);
    }

#pragma unroll
    for (int mt = 0; mt < 2; ++mt)
#pragma unroll
        for (int nt = 0; nt < 4; ++nt)
#pragma unroll
            for (int i = 0; i < 8; ++i) {
                int row = m_base + mt * 16 + i + 8 * half;
                int col = n_base + nt * 16 + r;
                size_t idx = (size_t)row * N + col;
                float v = acc[mt][nt][i];
                if (mode == 0) C[idx] = v;
                else           C[idx] += rowscale[row * NEXP + expert] * v;
            }
}

// ---------------------------------------------------------------------------
// Q/K flat RMSNorm + NeoX RoPE + bf16 pack; V bf16 pack. One block per token.
// ---------------------------------------------------------------------------
__global__ __launch_bounds__(256) void qk_prep(
    const float* __restrict__ qkv, const int* __restrict__ pos,
    const float* __restrict__ qnw, const float* __restrict__ knw,
    unsigned short* __restrict__ qb, unsigned short* __restrict__ kb,
    unsigned short* __restrict__ vb)
{
    __shared__ float redq[256], redk[256];
    const int t = blockIdx.x, tid = threadIdx.x;
    const float* base = qkv + (size_t)t * 3 * HIDDEN;
    float sq = 0.f, sk = 0.f;
    for (int i = tid; i < HIDDEN; i += 256) {
        float q = base[i];          sq += q * q;
        float k = base[HIDDEN + i]; sk += k * k;
    }
    redq[tid] = sq; redk[tid] = sk; __syncthreads();
    for (int s2 = 128; s2 > 0; s2 >>= 1) {
        if (tid < s2) { redq[tid] += redq[tid + s2]; redk[tid] += redk[tid + s2]; }
        __syncthreads();
    }
    const float rq = rsqrtf(redq[0] / HIDDEN + EPSV);
    const float rk = rsqrtf(redk[0] / HIDDEN + EPSV);
    const float p  = (float)pos[t];
    const float LNTH = 9.210340371976184f;     // ln(10000)
    for (int i = tid; i < HIDDEN; i += 256) {
        int d  = i & 127;
        int fi = d & 63;
        float ang = p * __expf(-(float)fi * (LNTH / 64.0f));
        float c = __cosf(ang), sn = __sinf(ang);
        int pidx = (d < 64) ? i + 64 : i - 64;
        float qn = base[i] * rq * qnw[i];
        float qp = base[pidx] * rq * qnw[pidx];
        float kn = base[HIDDEN + i] * rk * knw[i];
        float kp = base[HIDDEN + pidx] * rk * knw[pidx];
        float qo = (d < 64) ? qn * c - qp * sn : qn * c + qp * sn;
        float ko = (d < 64) ? kn * c - kp * sn : kn * c + kp * sn;
        qb[(size_t)t * HIDDEN + i] = f2bf(qo);
        kb[(size_t)t * HIDDEN + i] = f2bf(ko);
        vb[(size_t)t * HIDDEN + i] = f2bf(base[2 * HIDDEN + i]);
    }
}

// ---------------------------------------------------------------------------
// Causal flash attention, WMMA bf16. grid = (T/128, NH), block = 256 (8 waves).
// Each wave owns a 16-row Q tile. K tile is staged into LDS by the Tensor Data
// Mover (wave 0 issues a 2-group D# tensor_load_to_lds, TENSORcnt-tracked),
// overlapped with the all-thread V^T staging.
// ---------------------------------------------------------------------------
__global__ __launch_bounds__(256) void flash_attn(
    const unsigned short* __restrict__ Q, const unsigned short* __restrict__ Kb,
    const unsigned short* __restrict__ Vb, unsigned short* __restrict__ Ob)
{
    __shared__ __align__(16) unsigned short ksh[32 * 128];   // [key][d]  (TDM dest)
    __shared__ __align__(16) unsigned short vsh[128 * 32];   // [d][key]
    __shared__ __align__(16) unsigned short psh[8 * 16 * 32];

    const int tid  = threadIdx.x;
    const int wave = tid >> 5;
    const int lane = tid & 31;
    const int half = lane >> 4;
    const int r    = lane & 15;
    const int h    = blockIdx.y;
    const int q0   = blockIdx.x * 128 + wave * 16;
    const float scale = 0.08838834764831845f;   // 1/sqrt(128)

    // Wave-uniform TDM descriptor constants
    const unsigned int       lds_k = (unsigned int)(size_t)(&ksh[0]);
    const unsigned long long kbase =
        (unsigned long long)(size_t)Kb + (unsigned long long)h * HD * 2ull;

    Frag aq[4];
#pragma unroll
    for (int c = 0; c < 4; ++c)
        aq[c] = load2(Q + (size_t)(q0 + r) * HIDDEN + h * HD + c * 32 + half * 8);

    v8f o[8];
    float m[8], l[8];
#pragma unroll
    for (int nt = 0; nt < 8; ++nt)
#pragma unroll
        for (int i = 0; i < 8; ++i) o[nt][i] = 0.0f;
#pragma unroll
    for (int i = 0; i < 8; ++i) { m[i] = -1e30f; l[i] = 0.0f; }

    const int nkb = (blockIdx.x + 1) * 4;
    for (int kbi = 0; kbi < nkb; ++kbi) {
        const int kstart = kbi * 32;
        __syncthreads();   // previous iteration's LDS reads complete

        // --- TDM: DMA K tile (32 keys x 128 bf16) global -> ksh, wave 0 only ---
        if (wave == 0) {
            unsigned long long ga = kbase + (unsigned long long)kstart * (HIDDEN * 2);
            u32x4 g0;
            g0[0] = 1u;                                            // count=1, user mode
            g0[1] = lds_k;                                         // lds_addr
            g0[2] = (unsigned int)(ga & 0xffffffffu);              // global_addr[31:0]
            g0[3] = (unsigned int)((ga >> 32) & 0x1ffffffu)        // global_addr[56:32]
                  | (2u << 30);                                    // type=2 (image)
            u32x8 g1;
            g1[0] = (2u << 16);      // workgroup_mask=0, data_size=2 (4-byte units)
            g1[1] = (64u << 16);     // tensor_dim0 = 64 (x 4B = 128 bf16)
            g1[2] = (32u << 16);     // tensor_dim1 = 32 rows
            g1[3] = (64u << 16);     // tile_dim0 = 64
            g1[4] = 32u;             // tile_dim1 = 32, tile_dim2 = 0
            g1[5] = 1024u;           // tensor_dim0_stride = 1024 (x 4B = 2048 bf16)
            g1[6] = 0u;
            g1[7] = 0u;
            asm volatile("tensor_load_to_lds %0, %1" :: "s"(g0), "s"(g1) : "memory");
        }
        // --- all threads: stage V transposed [d][key] (overlaps with TDM) ---
        for (int i = tid; i < 4096; i += 256) {
            int d = i & 127, key = i >> 7;
            vsh[d * 32 + key] = Vb[(size_t)(kstart + key) * HIDDEN + h * HD + d];
        }
        if (wave == 0) __builtin_amdgcn_s_wait_tensorcnt(0);
        __syncthreads();
        if (kstart > q0 + 15) continue;     // fully masked for this wave

        // S = Q K^T for two 16-key tiles
        v8f s[2];
#pragma unroll
        for (int tk = 0; tk < 2; ++tk) {
            v8f sc;
#pragma unroll
            for (int i = 0; i < 8; ++i) sc[i] = 0.0f;
#pragma unroll
            for (int c = 0; c < 4; ++c) {
                Frag bk = load2(ksh + (tk * 16 + r) * 128 + c * 32 + half * 8);
                sc = wmma_bf16(aq[c].v, bk.v, sc);
            }
#pragma unroll
            for (int i = 0; i < 8; ++i) {
                int rowg = q0 + i + 8 * half;
                int colg = kstart + tk * 16 + r;
                s[tk][i] = (colg > rowg) ? -1e30f : sc[i] * scale;
            }
        }
        // online softmax (row reductions across 16-lane halves)
        float mnew[8], alpha[8];
#pragma unroll
        for (int i = 0; i < 8; ++i) {
            float rm = fmaxf(s[0][i], s[1][i]);
            for (int off = 1; off < 16; off <<= 1)
                rm = fmaxf(rm, __shfl_xor(rm, off, 16));
            mnew[i] = fmaxf(m[i], rm);
            alpha[i] = __expf(m[i] - mnew[i]);
            m[i] = mnew[i];
        }
#pragma unroll
        for (int i = 0; i < 8; ++i) {
            s[0][i] = __expf(s[0][i] - mnew[i]);
            s[1][i] = __expf(s[1][i] - mnew[i]);
            float rs = s[0][i] + s[1][i];
            for (int off = 1; off < 16; off <<= 1)
                rs += __shfl_xor(rs, off, 16);
            l[i] = l[i] * alpha[i] + rs;
        }
#pragma unroll
        for (int nt = 0; nt < 8; ++nt)
#pragma unroll
            for (int i = 0; i < 8; ++i) o[nt][i] *= alpha[i];

        // P through LDS: C-layout -> A-layout
        unsigned short* pw = psh + wave * 512;
#pragma unroll
        for (int tk = 0; tk < 2; ++tk)
#pragma unroll
            for (int i = 0; i < 8; ++i)
                pw[(i + 8 * half) * 32 + tk * 16 + r] = f2bf(s[tk][i]);
        Frag pa = load2(pw + r * 32 + half * 8);

        // O += P @ V
#pragma unroll
        for (int nt = 0; nt < 8; ++nt) {
            Frag bv = load2(vsh + (nt * 16 + r) * 32 + half * 8);
            o[nt] = wmma_bf16(pa.v, bv.v, o[nt]);
        }
    }

    float linv[8];
#pragma unroll
    for (int i = 0; i < 8; ++i) linv[i] = 1.0f / l[i];
#pragma unroll
    for (int nt = 0; nt < 8; ++nt)
#pragma unroll
        for (int i = 0; i < 8; ++i)
            Ob[(size_t)(q0 + i + 8 * half) * HIDDEN + h * HD + nt * 16 + r] =
                f2bf(o[nt][i] * linv[i]);
}

// ---------------------------------------------------------------------------
// out_f = rms_norm(xin)*w + res ; optional bf16 mirror. One block per token.
// ---------------------------------------------------------------------------
__global__ __launch_bounds__(256) void addnorm(
    const float* __restrict__ xin, const float* __restrict__ res,
    const float* __restrict__ w, float* __restrict__ out_f,
    unsigned short* __restrict__ out_b)
{
    __shared__ float red[256];
    const int t = blockIdx.x, tid = threadIdx.x;
    const float* x = xin + (size_t)t * HIDDEN;
    float s = 0.f;
    for (int i = tid; i < HIDDEN; i += 256) { float v = x[i]; s += v * v; }
    red[tid] = s; __syncthreads();
    for (int s2 = 128; s2 > 0; s2 >>= 1) {
        if (tid < s2) red[tid] += red[tid + s2];
        __syncthreads();
    }
    const float r = rsqrtf(red[0] / HIDDEN + EPSV);
    for (int i = tid; i < HIDDEN; i += 256) {
        float v = x[i] * r * w[i] + res[(size_t)t * HIDDEN + i];
        out_f[(size_t)t * HIDDEN + i] = v;
        if (out_b) out_b[(size_t)t * HIDDEN + i] = f2bf(v);
    }
}

// ---------------------------------------------------------------------------
// Router: wave-per-expert dot, then softmax + top-2 sparse combine weights.
// ---------------------------------------------------------------------------
__global__ __launch_bounds__(256) void router(
    const float* __restrict__ h, const float* __restrict__ gw,
    float* __restrict__ wdense)
{
    __shared__ float lg[8];
    const int t = blockIdx.x, tid = threadIdx.x, wave = tid >> 5, lane = tid & 31;
    const float* x = h + (size_t)t * HIDDEN;
    const float* g = gw + (size_t)wave * HIDDEN;
    float s = 0.f;
    for (int i = lane; i < HIDDEN; i += 32) s += x[i] * g[i];
    for (int off = 16; off > 0; off >>= 1) s += __shfl_xor(s, off, 32);
    if (lane == 0) lg[wave] = s;
    __syncthreads();
    if (tid == 0) {
        float mx = lg[0];
        for (int e = 1; e < NEXP; ++e) mx = fmaxf(mx, lg[e]);
        float pe[NEXP]; float sum = 0.f;
        for (int e = 0; e < NEXP; ++e) { pe[e] = __expf(lg[e] - mx); sum += pe[e]; }
        for (int e = 0; e < NEXP; ++e) pe[e] /= sum;
        int i1 = 0;
        for (int e = 1; e < NEXP; ++e) if (pe[e] > pe[i1]) i1 = e;    // first max (ties->low idx)
        int i2 = -1;
        for (int e = 0; e < NEXP; ++e) {
            if (e == i1) continue;
            if (i2 < 0 || pe[e] > pe[i2]) i2 = e;
        }
        for (int e = 0; e < NEXP; ++e)
            wdense[t * NEXP + e] = (e == i1 || e == i2) ? pe[e] : 0.0f;
    }
}

__global__ void cvt_bf16(const float* __restrict__ s, unsigned short* __restrict__ d, long n) {
    for (long i = blockIdx.x * (long)blockDim.x + threadIdx.x; i < n;
         i += (long)gridDim.x * blockDim.x)
        d[i] = f2bf(s[i]);
}

__global__ void zero_f32(float* __restrict__ d, long n) {
    for (long i = blockIdx.x * (long)blockDim.x + threadIdx.x; i < n;
         i += (long)gridDim.x * blockDim.x)
        d[i] = 0.0f;
}

__global__ void silumul(const float* __restrict__ g, const float* __restrict__ u,
                        unsigned short* __restrict__ hg, long n) {
    for (long i = blockIdx.x * (long)blockDim.x + threadIdx.x; i < n;
         i += (long)gridDim.x * blockDim.x) {
        float x = g[i];
        float s = x / (1.0f + __expf(-x));
        hg[i] = f2bf(s * u[i]);
    }
}

extern "C" void kernel_launch(void* const* d_in, const int* in_sizes, int n_in,
                              void* d_out, int out_size, void* d_ws, size_t ws_size,
                              hipStream_t stream)
{
    const int*   positions = (const int*)d_in[0];
    const float* x_in      = (const float*)d_in[1];
    const float* qkv_w     = (const float*)d_in[2];
    const float* o_w       = (const float*)d_in[3];
    const float* qnw       = (const float*)d_in[4];
    const float* knw       = (const float*)d_in[5];
    const float* panw      = (const float*)d_in[6];
    const float* pfnw      = (const float*)d_in[7];
    const float* gate_w    = (const float*)d_in[8];
    const float* w_gate    = (const float*)d_in[9];
    const float* w_up      = (const float*)d_in[10];
    const float* w_down    = (const float*)d_in[11];
    float* out = (float*)d_out;

    char* ws = (char*)d_ws;
    size_t off = 0;
    auto alloc = [&](size_t bytes) -> char* {
        char* p = ws + off;
        off += (bytes + 255) & ~(size_t)255;
        return p;
    };

    unsigned short* xb    = (unsigned short*)alloc((size_t)TT * HIDDEN * 2);
    unsigned short* qkvwb = (unsigned short*)alloc((size_t)3 * HIDDEN * HIDDEN * 2);
    unsigned short* owb   = (unsigned short*)alloc((size_t)HIDDEN * HIDDEN * 2);
    unsigned short* wgb   = (unsigned short*)alloc((size_t)NEXP * INTER * HIDDEN * 2);
    unsigned short* wub   = (unsigned short*)alloc((size_t)NEXP * INTER * HIDDEN * 2);
    unsigned short* wdb   = (unsigned short*)alloc((size_t)NEXP * HIDDEN * INTER * 2);
    float*          qkv_f = (float*)alloc((size_t)TT * 3 * HIDDEN * 4);
    unsigned short* qb    = (unsigned short*)alloc((size_t)TT * HIDDEN * 2);
    unsigned short* kb    = (unsigned short*)alloc((size_t)TT * HIDDEN * 2);
    unsigned short* vb    = (unsigned short*)alloc((size_t)TT * HIDDEN * 2);
    unsigned short* attnb = (unsigned short*)alloc((size_t)TT * HIDDEN * 2);
    float*          aproj = (float*)alloc((size_t)TT * HIDDEN * 4);
    float*          hbuf  = (float*)alloc((size_t)TT * HIDDEN * 4);
    unsigned short* hb    = (unsigned short*)alloc((size_t)TT * HIDDEN * 2);
    float*          wdns  = (float*)alloc((size_t)TT * NEXP * 4);
    float*          gbuf  = (float*)alloc((size_t)TT * INTER * 4);
    float*          ubuf  = (float*)alloc((size_t)TT * INTER * 4);
    unsigned short* hgb   = (unsigned short*)alloc((size_t)TT * INTER * 2);
    float*          moeb  = (float*)alloc((size_t)TT * HIDDEN * 4);

    // 1. fp32 -> bf16 conversions
    cvt_bf16<<<2048, 256, 0, stream>>>(x_in,   xb,    (long)TT * HIDDEN);
    cvt_bf16<<<2048, 256, 0, stream>>>(qkv_w,  qkvwb, (long)3 * HIDDEN * HIDDEN);
    cvt_bf16<<<2048, 256, 0, stream>>>(o_w,    owb,   (long)HIDDEN * HIDDEN);
    cvt_bf16<<<4096, 256, 0, stream>>>(w_gate, wgb,   (long)NEXP * INTER * HIDDEN);
    cvt_bf16<<<4096, 256, 0, stream>>>(w_up,   wub,   (long)NEXP * INTER * HIDDEN);
    cvt_bf16<<<4096, 256, 0, stream>>>(w_down, wdb,   (long)NEXP * HIDDEN * INTER);

    // 2. QKV projection
    gemm_bf16<<<dim3(3 * HIDDEN / 128, TT / 128), 256, 0, stream>>>(
        xb, qkvwb, qkv_f, TT, 3 * HIDDEN, HIDDEN, 0, nullptr, 0);

    // 3. q/k norm + RoPE + bf16 pack
    qk_prep<<<TT, 256, 0, stream>>>(qkv_f, positions, qnw, knw, qb, kb, vb);

    // 4. flash attention (TDM-staged K tiles)
    flash_attn<<<dim3(TT / 128, NH), 256, 0, stream>>>(qb, kb, vb, attnb);

    // 5. output projection
    gemm_bf16<<<dim3(HIDDEN / 128, TT / 128), 256, 0, stream>>>(
        attnb, owb, aproj, TT, HIDDEN, HIDDEN, 0, nullptr, 0);

    // 6. post-attn norm + residual (fp32 h and bf16 mirror)
    addnorm<<<TT, 256, 0, stream>>>(aproj, x_in, panw, hbuf, hb);

    // 7. router + top-2 combine weights
    router<<<TT, 256, 0, stream>>>(hbuf, gate_w, wdns);

    // 8. MoE
    zero_f32<<<2048, 256, 0, stream>>>(moeb, (long)TT * HIDDEN);
    for (int e = 0; e < NEXP; ++e) {
        gemm_bf16<<<dim3(INTER / 128, TT / 128), 256, 0, stream>>>(
            hb, wgb + (size_t)e * INTER * HIDDEN, gbuf, TT, INTER, HIDDEN, 0, nullptr, 0);
        gemm_bf16<<<dim3(INTER / 128, TT / 128), 256, 0, stream>>>(
            hb, wub + (size_t)e * INTER * HIDDEN, ubuf, TT, INTER, HIDDEN, 0, nullptr, 0);
        silumul<<<2048, 256, 0, stream>>>(gbuf, ubuf, hgb, (long)TT * INTER);
        gemm_bf16<<<dim3(HIDDEN / 128, TT / 128), 256, 0, stream>>>(
            hgb, wdb + (size_t)e * HIDDEN * INTER, moeb, TT, HIDDEN, INTER, 1, wdns, e);
    }

    // 9. final norm + residual -> out
    addnorm<<<TT, 256, 0, stream>>>(moeb, hbuf, pfnw, out, nullptr);
}
